// SlowFluidNet_83863531422265
// MI455X (gfx1250) — matrix operand order repaired and checked by reading, not compiled, not run on hardware
//
#include <hip/hip_runtime.h>
#include <hip/hip_bf16.h>

// ---------------------------------------------------------------------------
// SlowFluidNet for MI455X (gfx1250, wave32, WMMA).
//
// Layer-1 preactivation separates as A[n] + B[p] (precomputed, stored with the
// 18 features SPLIT across both wave halves: feats 0..8 at K=0..8, feats 9..17
// at K=16..24, row stride 32 floats). The 18->9->6->3 tails run on
// v_wmma_f32_16x16x32_f16, 16 pairs per wave-step:
//   - layer 2: two WMMAs (fluid, solid). The solid A fragment's output rows
//     are ROTATED (feature f -> row (f+8)&15) so solid h2 lands in lanes>=16.
//   - layers 3+4: fluid and solid stacked into one WMMA each (fluid block at
//     M=0../K=0.., solid block at M=8../K=16..). Thanks to the rotation, the
//     stacked layer-3 B operand needs only TWO ds_bpermute pulls (feature 8
//     of each MLP); layer 4 and the masked accumulation need none.
//   - B-fragment padding elements are written as exact zeros, and the matching A elements
//     are exact zeros as well.
// Activations use the hardware transcendental v_tanh_f32.
// ---------------------------------------------------------------------------

typedef __attribute__((ext_vector_type(16))) _Float16 v16h;
typedef __attribute__((ext_vector_type(8)))  float    v8f;

#define ROWF 32  // floats per padded layer-1 row (128 B)

__device__ __forceinline__ float tanh_hw(float x) {
#if __has_builtin(__builtin_amdgcn_tanhf)
  return __builtin_amdgcn_tanhf(x);
#else
  float r;
  asm volatile("v_tanh_f32 %0, %1" : "=v"(r) : "v"(x));
  return r;
#endif
}

// feature k (0..17) -> padded slot: 0..8 stay, 9..17 go to 16..24.
__device__ __host__ __forceinline__ int slot18(int k) {
  return (k < 9) ? k : (16 + (k - 9));
}

// ---- Stage 1: per-neighbor layer-1 partials ------------------------------
__global__ void fluidnet_stage_n(const float* __restrict__ data,
                                 const float* __restrict__ fw0,
                                 const float* __restrict__ sw0,
                                 float* __restrict__ Af,
                                 float* __restrict__ As,
                                 float* __restrict__ flagf,
                                 int N) {
  int n = blockIdx.x * blockDim.x + threadIdx.x;
  if (n >= N) return;
  float d0 = data[n * 7 + 0], d1 = data[n * 7 + 1], d2 = data[n * 7 + 2];
  float d3 = data[n * 7 + 3], d4 = data[n * 7 + 4], d5 = data[n * 7 + 5];
  flagf[n] = data[n * 7 + 6];
  float* afr = Af + (size_t)n * ROWF;
  float* asr = As + (size_t)n * ROWF;
  for (int s = 0; s < ROWF; ++s) { afr[s] = 0.f; asr[s] = 0.f; }
  for (int k = 0; k < 18; ++k) {
    // fluid_in = [data[:3]-cen[:3], data[3:6], cen[3:6]] -> data part rows 0..5
    float af = d0 * fw0[k] + d1 * fw0[18 + k] + d2 * fw0[36 + k] +
               d3 * fw0[54 + k] + d4 * fw0[72 + k] + d5 * fw0[90 + k];
    // solid_in = [data[:3]-cen[:3], cen[3:6]] -> data part rows 0..2
    float as_ = d0 * sw0[k] + d1 * sw0[18 + k] + d2 * sw0[36 + k];
    afr[slot18(k)] = af;
    asr[slot18(k)] = as_;
  }
}

// ---- Stage 2: per-particle layer-1 partials (bias folded in) -------------
__global__ void fluidnet_stage_p(const float* __restrict__ centers,
                                 const float* __restrict__ fw0,
                                 const float* __restrict__ fb0,
                                 const float* __restrict__ sw0,
                                 const float* __restrict__ sb0,
                                 float* __restrict__ Bf,
                                 float* __restrict__ Bs,
                                 int P) {
  int p = blockIdx.x * blockDim.x + threadIdx.x;
  if (p >= P) return;
  float c0 = centers[p * 6 + 0], c1 = centers[p * 6 + 1], c2 = centers[p * 6 + 2];
  float c3 = centers[p * 6 + 3], c4 = centers[p * 6 + 4], c5 = centers[p * 6 + 5];
  float* bfr = Bf + (size_t)p * ROWF;
  float* bsr = Bs + (size_t)p * ROWF;
  for (int s = 0; s < ROWF; ++s) { bfr[s] = 0.f; bsr[s] = 0.f; }
  for (int k = 0; k < 18; ++k) {
    float bf = -c0 * fw0[k] - c1 * fw0[18 + k] - c2 * fw0[36 + k]
               + c3 * fw0[108 + k] + c4 * fw0[126 + k] + c5 * fw0[144 + k] + fb0[k];
    float bs = -c0 * sw0[k] - c1 * sw0[18 + k] - c2 * sw0[36 + k]
               + c3 * sw0[54 + k] + c4 * sw0[72 + k] + c5 * sw0[90 + k] + sb0[k];
    bfr[slot18(k)] = bf;
    bsr[slot18(k)] = bs;
  }
}

// ---- WMMA fragment builders ----------------------------------------------
// A operand element t -> K = 16*hs + t; lane row M = lane&15.

// Layer-2 A with split-K input mapping (K<9 -> feat K; 16<=K<25 -> feat K-7).
// rot=0: fluid (feature f at row f). rot=1: solid (feature f at row (f+8)&15).
__device__ __forceinline__ v16h load_afrag_l1(const float* __restrict__ W,
                                              int out, int q, int hs, int rot) {
  int f = rot ? ((q + 8) & 15) : q;
  v16h a;
#pragma unroll
  for (int t = 0; t < 16; ++t) {
    int K = 16 * hs + t;
    bool kv = (K < 9) || (K >= 16 && K < 25);
    int kk = (K < 16) ? K : (K - 7);
    float v = (kv && f < out) ? W[kk * out + f] : 0.0f;
    a[t] = (_Float16)v;
  }
  return a;
}

// Stacked A: fluid block M=0..out-1 / K=0..in-1, solid block M=8.. / K=16..
__device__ __forceinline__ v16h load_afrag2(const float* __restrict__ Wf,
                                            const float* __restrict__ Ws,
                                            int in, int out, int q, int hs) {
  v16h a;
#pragma unroll
  for (int t = 0; t < 16; ++t) {
    int K = 16 * hs + t;
    float v = 0.0f;
    if (q < out && K < in)                           v = Wf[K * out + q];
    else if (q >= 8 && (q - 8) < out && K >= 16 && (K - 16) < in)
                                                     v = Ws[(K - 16) * out + (q - 8)];
    a[t] = (_Float16)v;
  }
  return a;
}

// C operand: lane<16 VGPR r -> M=r; lane>=16 -> M=r+8.
// rot variant matches the rotated solid layer-2 rows.
__device__ __forceinline__ v8f load_cfrag_rot(const float* __restrict__ b,
                                              int out, int hs, int rot) {
  v8f c;
#pragma unroll
  for (int r = 0; r < 8; ++r) {
    int M = r + 8 * hs;
    int f = rot ? ((M + 8) & 15) : M;
    c[r] = (f < out) ? b[f] : 0.0f;
  }
  return c;
}

// Stacked biases: fluid at M=0..out-1, solid at M=8..8+out-1.
__device__ __forceinline__ v8f load_cfrag2(const float* __restrict__ bf,
                                           const float* __restrict__ bs,
                                           int out, int hs) {
  v8f c;
#pragma unroll
  for (int r = 0; r < 8; ++r) {
    int M = r + 8 * hs;
    float v = 0.0f;
    if (M < out)                      v = bf[M];
    else if (M >= 8 && (M - 8) < out) v = bs[M - 8];
    c[r] = v;
  }
  return c;
}

// Load 12 floats (features 0..8 of this lane's K-chunk + zero pad).
__device__ __forceinline__ void load_row12(const float* __restrict__ base,
                                           float* __restrict__ dst) {
  const float4* f4 = reinterpret_cast<const float4*>(base);
  float4 a = f4[0], b = f4[1], c = f4[2];
  dst[0] = a.x; dst[1] = a.y; dst[2]  = a.z; dst[3]  = a.w;
  dst[4] = b.x; dst[5] = b.y; dst[6]  = b.z; dst[7]  = b.w;
  dst[8] = c.x; dst[9] = c.y; dst[10] = c.z; dst[11] = c.w;
}

// ---- Stage 3: main pairwise kernel, one workgroup per particle -----------
__global__ __launch_bounds__(256) void fluidnet_main(
    const unsigned char* __restrict__ mask,
    const float* __restrict__ Af, const float* __restrict__ As,
    const float* __restrict__ Bf, const float* __restrict__ Bs,
    const float* __restrict__ flagf,
    const float* __restrict__ fw1, const float* __restrict__ fb1,
    const float* __restrict__ fw2, const float* __restrict__ fb2,
    const float* __restrict__ fw3, const float* __restrict__ fb3,
    const float* __restrict__ sw1, const float* __restrict__ sb1,
    const float* __restrict__ sw2, const float* __restrict__ sb2,
    const float* __restrict__ sw3, const float* __restrict__ sb3,
    float* __restrict__ out, int N) {
  __shared__ float s_out[3];
  const int p = blockIdx.x;
  const int tid = threadIdx.x;
  if (tid < 3) s_out[tid] = 0.0f;
  __syncthreads();

  const int lane = tid & 31;
  const int wv = tid >> 5;
  const int hs = lane >> 4;
  const int q = lane & 15;
  const int idx_lo = (lane & 15) << 2;  // bpermute: pull from low half
  const int idx_hi = (lane | 16) << 2;  // bpermute: pull from high half

  // Weight / bias fragments, resident in VGPRs for the whole kernel.
  v16h aF1 = load_afrag_l1(fw1, 9, q, hs, /*rot=*/0);
  v16h aS1 = load_afrag_l1(sw1, 9, q, hs, /*rot=*/1);
  v16h a2  = load_afrag2(fw2, sw2, 9, 6, q, hs);  // stacked 9->6
  v16h a3  = load_afrag2(fw3, sw3, 6, 3, q, hs);  // stacked 6->3
  v8f cF1 = load_cfrag_rot(fb1, 9, hs, 0);
  v8f cS1 = load_cfrag_rot(sb1, 9, hs, 1);
  v8f c2  = load_cfrag2(fb2, sb2, 6, hs);
  v8f c3  = load_cfrag2(fb3, sb3, 3, hs);

  // Per-particle layer-1 partials for this lane's K-chunk (9 live values).
  float bvF[12], bvS[12];
  load_row12(Bf + (size_t)p * ROWF + 16 * hs, bvF);
  load_row12(Bs + (size_t)p * ROWF + 16 * hs, bvS);

  float acc0 = 0.0f, acc1 = 0.0f, acc2 = 0.0f;

  const int ntiles = N >> 4;  // 16 pairs per wave-step
  for (int tile = wv; tile < ntiles; tile += 8) {
    const int n = (tile << 4) + q;
    float av[12], sv[12];
    load_row12(Af + (size_t)n * ROWF + 16 * hs, av);
    load_row12(As + (size_t)n * ROWF + 16 * hs, sv);
    const unsigned char mb = mask[(size_t)p * N + n];
    const float fl = flagf[n];
    const float wF = (mb && fl < 0.5f) ? 1.0f : 0.0f;
    const float wS = (mb && fl >= 0.5f) ? 1.0f : 0.0f;

    // Layer-2 inputs: h1 = tanh(A[n] + B[p]). Split-K layout makes every
    // lane live for t=0..8; t=9..15 multiply exact-zero A columns.
    v16h b1f, b1s;
#pragma unroll
    for (int t = 0; t < 9; ++t) {
      b1f[t] = (_Float16)tanh_hw(av[t] + bvF[t]);
      b1s[t] = (_Float16)tanh_hw(sv[t] + bvS[t]);
    }
#pragma unroll
    for (int t = 9; t < 16; ++t) {
      b1f[t] = (_Float16)0.0f;
      b1s[t] = (_Float16)0.0f;
    }
    v8f d2f = __builtin_amdgcn_wmma_f32_16x16x32_f16(false, aF1, false, b1f,
                                                     (short)0, cF1, false, false);
    v8f d2s = __builtin_amdgcn_wmma_f32_16x16x32_f16(false, aS1, false, b1s,
                                                     (short)0, cS1, false, false);

    // h2 = tanh(d2). Fluid: feats 0..7 in lanes<16, feat 8 in lanes>=16 (r=0).
    // Solid (rotated): feats 0..7 in lanes>=16, feat 8 in lanes<16 (r=0).
    float tf[8], ts[8];
#pragma unroll
    for (int r = 0; r < 8; ++r) {
      tf[r] = tanh_hw(d2f[r]);
      ts[r] = tanh_hw(d2s[r]);
    }
    // Stacked layer-3 B: fluid h2 at K=0..8 (lanes<16), solid h2 at K=16..24
    // (lanes>=16). Only feature 8 of each MLP crosses halves.
    float f8f = __int_as_float(
        __builtin_amdgcn_ds_bpermute(idx_hi, __float_as_int(tf[0])));
    float f8s = __int_as_float(
        __builtin_amdgcn_ds_bpermute(idx_lo, __float_as_int(ts[0])));
    v16h b2;
#pragma unroll
    for (int t = 0; t < 8; ++t) b2[t] = (_Float16)(hs ? ts[t] : tf[t]);
    b2[8] = (_Float16)(hs ? f8s : f8f);
#pragma unroll
    for (int t = 9; t < 16; ++t) b2[t] = (_Float16)0.0f;

    v8f d3 = __builtin_amdgcn_wmma_f32_16x16x32_f16(false, a2, false, b2,
                                                    (short)0, c2, false, false);

    // Stacked layer-4 B: lanes<16 natively hold fluid h3 (rows 0..5),
    // lanes>=16 natively hold solid h3 (rows 8..13). No cross-lane traffic.
    v16h b3;
#pragma unroll
    for (int t = 0; t < 6; ++t) b3[t] = (_Float16)tanh_hw(d3[t]);
#pragma unroll
    for (int t = 6; t < 16; ++t) b3[t] = (_Float16)0.0f;

    v8f d4 = __builtin_amdgcn_wmma_f32_16x16x32_f16(false, a3, false, b3,
                                                    (short)0, c3, false, false);

    // lanes<16: fluid outputs (rows 0..2); lanes>=16: solid outputs (rows 8..10).
    const float w = hs ? wS : wF;
    acc0 += w * d4[0];
    acc1 += w * d4[1];
    acc2 += w * d4[2];
  }

  atomicAdd(&s_out[0], acc0);
  atomicAdd(&s_out[1], acc1);
  atomicAdd(&s_out[2], acc2);
  __syncthreads();
  if (tid < 3) out[p * 3 + tid] = s_out[tid];
}

// ---------------------------------------------------------------------------
extern "C" void kernel_launch(void* const* d_in, const int* in_sizes, int n_in,
                              void* d_out, int out_size, void* d_ws, size_t ws_size,
                              hipStream_t stream) {
  const unsigned char* mask = (const unsigned char*)d_in[0];
  const float* centers = (const float*)d_in[1];
  const float* data    = (const float*)d_in[2];
  const float* fw0 = (const float*)d_in[3];
  const float* fb0 = (const float*)d_in[4];
  const float* sw0 = (const float*)d_in[5];
  const float* sb0 = (const float*)d_in[6];
  const float* fw1 = (const float*)d_in[7];
  const float* fb1 = (const float*)d_in[8];
  const float* sw1 = (const float*)d_in[9];
  const float* sb1 = (const float*)d_in[10];
  const float* fw2 = (const float*)d_in[11];
  const float* fb2 = (const float*)d_in[12];
  const float* sw2 = (const float*)d_in[13];
  const float* sb2 = (const float*)d_in[14];
  const float* fw3 = (const float*)d_in[15];
  const float* fb3 = (const float*)d_in[16];
  const float* sw3 = (const float*)d_in[17];
  const float* sb3 = (const float*)d_in[18];

  const int P = in_sizes[1] / 6;
  const int N = in_sizes[2] / 7;

  float* wsf = (float*)d_ws;
  float* Af    = wsf;                         // N*ROWF
  float* As    = Af + (size_t)N * ROWF;       // N*ROWF
  float* Bf    = As + (size_t)N * ROWF;       // P*ROWF
  float* Bs    = Bf + (size_t)P * ROWF;       // P*ROWF
  float* flagf = Bs + (size_t)P * ROWF;       // N

  fluidnet_stage_n<<<(N + 255) / 256, 256, 0, stream>>>(data, fw0, sw0,
                                                        Af, As, flagf, N);
  fluidnet_stage_p<<<(P + 255) / 256, 256, 0, stream>>>(centers, fw0, fb0,
                                                        sw0, sb0, Bf, Bs, P);
  fluidnet_main<<<P, 256, 0, stream>>>(mask, Af, As, Bf, Bs, flagf,
                                       fw1, fb1, fw2, fb2, fw3, fb3,
                                       sw1, sb1, sw2, sb2, sw3, sb3,
                                       (float*)d_out, N);
}
